// FFTAttention_73186242724050
// MI455X (gfx1250) — matrix operand order repaired and checked by reading, not compiled, not run on hardware
//
#include <hip/hip_runtime.h>
#include <hip/hip_bf16.h>
#include <math.h>

#define HIDDEN 1024
#define HEADS 16
#define HEAD_DIM 64
#define BATCH 4
#define SEQ 4096
#define ROWS (BATCH * SEQ) // 16384

typedef __attribute__((ext_vector_type(16))) __bf16 v16bf;
typedef __attribute__((ext_vector_type(8)))  float  v8f;
typedef int v4i __attribute__((__vector_size__(16))); // matches async builtin param type

#define GAS __attribute__((address_space(1)))
#define LAS __attribute__((address_space(3)))

#if defined(__has_builtin)
#if __has_builtin(__builtin_amdgcn_global_load_async_to_lds_b128)
#define HAVE_ASYNC_LDS 1
#endif
#endif

static __device__ __forceinline__ unsigned short f2bf(float f) {
    unsigned u = __float_as_uint(f);
    unsigned r = (u + 0x7FFFu + ((u >> 16) & 1u)) >> 16;
    return (unsigned short)r;
}

template <int N>
static __device__ __forceinline__ void wait_async_lds_n() {
#if defined(HAVE_ASYNC_LDS)
#if __has_builtin(__builtin_amdgcn_s_wait_asynccnt)
    __builtin_amdgcn_s_wait_asynccnt(N);
#else
    asm volatile("s_wait_asynccnt %0" ::"i"(N) : "memory");
#endif
#endif
}

// ---------------------------------------------------------------------------
// fp32 -> bf16 conversion of activations
// ---------------------------------------------------------------------------
__global__ void cvt_x_kernel(const float* __restrict__ x, unsigned short* __restrict__ xb) {
    int i = blockIdx.x * 256 + threadIdx.x;
    xb[i] = f2bf(x[i]);
}

// ---------------------------------------------------------------------------
// Convert the four weight matrices into tile-blocked B layout:
// block (nt, kt) holds the exact 128(N) x 32(K) LDS tile, N-major:
//   Wblk[((nt*32+kt)*128 + n)*32 + k] = W[(nt*128+n)][kt*32+k]   (= B[k][n])
// so GEMM B-staging is contiguous 16B copies (async-DMA friendly).
// ---------------------------------------------------------------------------
__global__ void cvt_w_kernel(const float* __restrict__ Wq, const float* __restrict__ Wk,
                             const float* __restrict__ Wv, const float* __restrict__ Wo,
                             unsigned short* __restrict__ Wblk) {
    int wi = blockIdx.y;
    const float* W = (wi == 0) ? Wq : (wi == 1) ? Wk : (wi == 2) ? Wv : Wo;
    unsigned short* dst = Wblk + (size_t)wi * HIDDEN * HIDDEN;
    int idx = blockIdx.x * 256 + threadIdx.x; // over 1M elements
    int block = idx >> 12;      // /(128*32)
    int n = (idx >> 5) & 127;
    int k = idx & 31;
    int nt = block >> 5, kt = block & 31;
    int ng = nt * 128 + n, kg = kt * 32 + k;
    dst[idx] = f2bf(W[(size_t)ng * HIDDEN + kg]);
}

// ---------------------------------------------------------------------------
// WMMA bf16 GEMM: Y[row, col] = sum_k X[row,k] * W^T[k,col] + bias[col]
// Workgroup = 256 threads (8 wave32) computes a 128(M) x 128(N) tile;
// waves arranged 4(M) x 2(N), each wave: 2 A-frags x 4 B-frags = 8 WMMA / K-slice.
// A and B both staged via global_load_async_to_lds_b128 with LDS double buffer.
// MODE 0: three weight mats (blockIdx.z selects q/k/v), scatter to [B,H*D,S] fp32
// MODE 1: single weight mat, plain fp32 row-major output (final projection)
// ---------------------------------------------------------------------------
template <int MODE>
__global__ __launch_bounds__(256) void gemm_wmma_kernel(
    const unsigned short* __restrict__ X,  // [ROWS][HIDDEN] bf16
    const unsigned short* __restrict__ Wb, // blocked bf16 (x3 for MODE 0)
    const float* __restrict__ b0, const float* __restrict__ b1, const float* __restrict__ b2,
    float* __restrict__ o0, float* __restrict__ o1, float* __restrict__ o2) {

    const int lane = threadIdx.x & 31;
    const int wave = threadIdx.x >> 5;
    const int waveM = wave >> 1; // 0..3 -> 32-row slab
    const int waveN = wave & 1;  // 0..1 -> 64-col slab
    const int row0 = blockIdx.y * 128;
    const int col0 = blockIdx.x * 128;

    const unsigned short* W = Wb;
    const float* bias = b0;
    float* out = o0;
    if (MODE == 0) {
        int z = blockIdx.z;
        W = Wb + (size_t)z * HIDDEN * HIDDEN;
        bias = (z == 0) ? b0 : (z == 1) ? b1 : b2;
        out  = (z == 0) ? o0 : (z == 1) ? o1 : o2;
    }

    // 80-byte row stride => 16B-aligned b128 fragment loads, conflict-spread
    __shared__ __align__(16) unsigned short As[2][128][40];
    __shared__ __align__(16) unsigned short Bs[2][128][40];

    v8f acc[2][4];
#pragma unroll
    for (int mi = 0; mi < 2; mi++)
#pragma unroll
        for (int ni = 0; ni < 4; ni++) acc[mi][ni] = {};

    union Frag { v16bf v; uint4 q[2]; };
    Frag fa[2], fb;

    const unsigned* Xu = (const unsigned*)X;
    const unsigned* Wu = (const unsigned*)W;

    // Stage one 128x32 A-tile + 128x32 B-tile into LDS buffer `buf`
    // (4 async b128 chunks per thread; 512 chunks per tile over 256 threads).
    auto stage = [&](int buf, int kks) {
#if defined(HAVE_ASYNC_LDS)
#pragma unroll
        for (int i = 0; i < 2; i++) {
            int e = threadIdx.x + i * 256; // < 512
            int r = e >> 2, c4 = e & 3;
            __builtin_amdgcn_global_load_async_to_lds_b128(
                (GAS v4i*)(Xu + (size_t)(row0 + r) * 512 + (kks >> 1) + c4 * 4),
                (LAS v4i*)&As[buf][r][c4 * 8], 0, 0);
        }
        const unsigned* Wt = Wu + ((size_t)(col0 >> 7) * 32 + (kks >> 5)) * 2048; // block base (uints)
#pragma unroll
        for (int i = 0; i < 2; i++) {
            int e = threadIdx.x + i * 256; // < 512
            int n = e >> 2, c4 = e & 3;
            __builtin_amdgcn_global_load_async_to_lds_b128(
                (GAS v4i*)(Wt + (size_t)n * 16 + c4 * 4),
                (LAS v4i*)&Bs[buf][n][c4 * 8], 0, 0);
        }
#else
#pragma unroll
        for (int i = 0; i < 8; i++) {
            int e = threadIdx.x + i * 256; // < 2048
            int r = e >> 4, c2 = e & 15;
            unsigned u = Xu[(size_t)(row0 + r) * 512 + (kks >> 1) + c2];
            *(unsigned*)&As[buf][r][c2 * 2] = u;
        }
        const unsigned* Wt = Wu + ((size_t)(col0 >> 7) * 32 + (kks >> 5)) * 2048;
#pragma unroll
        for (int i = 0; i < 8; i++) {
            int e = threadIdx.x + i * 256; // < 2048
            int n = e >> 4, c2 = e & 15;
            unsigned u = Wt[(size_t)n * 16 + c2];
            *(unsigned*)&Bs[buf][n][c2 * 2] = u;
        }
#endif
    };

    auto compute = [&](int buf) {
        // A fragments (16-bit ISA layout: lanes 0-15 M=0..15 hold K {b..b+7, 16+b..})
        int abase = (lane >> 4) * 8;
#pragma unroll
        for (int mi = 0; mi < 2; mi++) {
            int aM = waveM * 32 + mi * 16 + (lane & 15);
            fa[mi].q[0] = *(const uint4*)&As[buf][aM][abase];
            fa[mi].q[1] = *(const uint4*)&As[buf][aM][16 + abase];
        }
        int klo = (lane >> 4) * 8; // B: lanes 0-15 K={0..7,16..23}, lanes 16-31 K={8..15,24..31}
#pragma unroll
        for (int ni = 0; ni < 4; ni++) {
            int bn = waveN * 64 + ni * 16 + (lane & 15);
            fb.q[0] = *(const uint4*)&Bs[buf][bn][klo];
            fb.q[1] = *(const uint4*)&Bs[buf][bn][klo + 16];
#pragma unroll
            for (int mi = 0; mi < 2; mi++) {
                acc[mi][ni] = __builtin_amdgcn_wmma_f32_16x16x32_bf16(
                    false, fa[mi].v, false, fb.v, (short)0, acc[mi][ni], false, false);
            }
        }
    };

#if defined(HAVE_ASYNC_LDS)
    // Software pipeline: DMA for slice t+1 overlaps WMMA on slice t.
    stage(0, 0);
    for (int kk = 0; kk < HIDDEN; kk += 32) {
        int cur = (kk >> 5) & 1;
        if (kk + 32 < HIDDEN) {
            stage(cur ^ 1, kk + 32);
            __builtin_prefetch(&X[(size_t)(row0 + (threadIdx.x >> 1)) * HIDDEN + kk + 64], 0, 0);
            wait_async_lds_n<4>(); // own 4 chunks of `cur` done (async completes in order)
        } else {
            wait_async_lds_n<0>();
        }
        __syncthreads(); // all waves' `cur` staging complete
        compute(cur);
        __syncthreads(); // all reads of `cur` done before it is overwritten
    }
#else
    for (int kk = 0; kk < HIDDEN; kk += 32) {
        stage(0, kk);
        __syncthreads();
        compute(0);
        __syncthreads();
    }
#endif

    // Epilogue: C/D layout: lanes 0-15 N=lane, M=vgpr; lanes 16-31 N=lane-16, M=8+vgpr
#pragma unroll
    for (int mi = 0; mi < 2; mi++) {
#pragma unroll
        for (int ni = 0; ni < 4; ni++) {
#pragma unroll
            for (int r = 0; r < 8; r++) {
                int M = (lane >> 4) * 8 + r;
                int N = lane & 15;
                int row = row0 + waveM * 32 + mi * 16 + M;
                int col = col0 + waveN * 64 + ni * 16 + N;
                float val = acc[mi][ni][r] + bias[col];
                if (MODE == 0) {
                    int bb = row >> 12, s = row & 4095;
                    // channel-major [B, H*D, S] so FFT loads are s-contiguous
                    out[((size_t)(bb * HIDDEN + col)) * SEQ + s] = val;
                } else {
                    out[(size_t)row * HIDDEN + col] = val;
                }
            }
        }
    }
}

// ---------------------------------------------------------------------------
// In-LDS radix-2 complex FFT of length 4096 (256 threads, 16 pts/thread).
// sign = -1 forward, +1 inverse (caller normalizes by 1/N).
// ---------------------------------------------------------------------------
static __device__ __forceinline__ void fft4096_inplace(float2* w, int tid, float sign) {
#pragma unroll
    for (int i = 0; i < 16; i++) { // bit-reverse permutation (12 bits)
        int idx = tid + i * 256;
        int r = (int)(__brev((unsigned)idx) >> 20);
        if (idx < r) { float2 t = w[idx]; w[idx] = w[r]; w[r] = t; }
    }
    __syncthreads();
    for (int len = 2; len <= 4096; len <<= 1) {
        int half = len >> 1;
        float base = sign * 6.283185307179586f / (float)len;
        for (int j = tid; j < 2048; j += 256) {
            int blk = j / half;
            int pos = j - blk * half;
            int i0 = blk * len + pos;
            int i1 = i0 + half;
            float sn, cs;
            __sincosf(base * (float)pos, &sn, &cs);
            float2 a = w[i0], b = w[i1];
            float tr = b.x * cs - b.y * sn;
            float ti = b.x * sn + b.y * cs;
            w[i0] = make_float2(a.x + tr, a.y + ti);
            w[i1] = make_float2(a.x - tr, a.y - ti);
        }
        __syncthreads();
    }
}

// ---------------------------------------------------------------------------
// One workgroup per channel ch = b*1024 + (h*64+d); q/k/v are [B*C][S] fp32,
// so every global access is s-contiguous (fully coalesced).
// out = Re(IFFT(FFT(q) * conj(FFT(k)) * FFT(v))) / N / sqrt(head_dim),
// written bf16 channel-major to outc[ch][s].
// ---------------------------------------------------------------------------
__global__ __launch_bounds__(256) void fftconv_kernel(
    const float* __restrict__ q, const float* __restrict__ k, const float* __restrict__ v,
    unsigned short* __restrict__ outc) {
    __shared__ float2 work[4096]; // 32 KB
    int ch = blockIdx.x; // 0..4095
    int tid = threadIdx.x;
    size_t base = (size_t)ch * SEQ;

    float qr[16], qi[16], ar[16], ai[16];

    for (int i = 0; i < 16; i++) { // FFT(q)
        int s = tid + i * 256;
        work[s] = make_float2(q[base + s], 0.f);
    }
    __syncthreads();
    fft4096_inplace(work, tid, -1.f);
    for (int i = 0; i < 16; i++) { int s = tid + i * 256; qr[i] = work[s].x; qi[i] = work[s].y; }
    __syncthreads();

    for (int i = 0; i < 16; i++) { // FFT(k); a = q * conj(k)
        int s = tid + i * 256;
        work[s] = make_float2(k[base + s], 0.f);
    }
    __syncthreads();
    fft4096_inplace(work, tid, -1.f);
    for (int i = 0; i < 16; i++) {
        int s = tid + i * 256;
        float kr = work[s].x, ki = work[s].y;
        ar[i] = qr[i] * kr + qi[i] * ki;
        ai[i] = qi[i] * kr - qr[i] * ki;
    }
    __syncthreads();

    for (int i = 0; i < 16; i++) { // FFT(v); p = a * v
        int s = tid + i * 256;
        work[s] = make_float2(v[base + s], 0.f);
    }
    __syncthreads();
    fft4096_inplace(work, tid, -1.f);
    for (int i = 0; i < 16; i++) {
        int s = tid + i * 256;
        float vr = work[s].x, vi = work[s].y;
        work[s] = make_float2(ar[i] * vr - ai[i] * vi, ar[i] * vi + ai[i] * vr);
    }
    __syncthreads();

    fft4096_inplace(work, tid, +1.f); // inverse FFT
    const float norm = (1.0f / 4096.0f) * 0.125f; // 1/N * 1/sqrt(64)
    for (int i = 0; i < 16; i++) {
        int s = tid + i * 256;
        outc[base + s] = f2bf(work[s].x * norm); // coalesced bf16 store
    }
}

// ---------------------------------------------------------------------------
// LDS-tiled transpose: outc [B][C=1024][S=4096] bf16 -> attnb [B][S][C] bf16.
// 32x32 tiles, coalesced on both sides.
// ---------------------------------------------------------------------------
__global__ __launch_bounds__(256) void transpose_kernel(const unsigned short* __restrict__ src,
                                                        unsigned short* __restrict__ dst) {
    __shared__ unsigned short tile[32][33];
    int b = blockIdx.z;
    int s0 = blockIdx.x * 32;
    int c0 = blockIdx.y * 32;
    for (int i = threadIdx.y; i < 32; i += 8)
        tile[i][threadIdx.x] = src[((size_t)b * HIDDEN + c0 + i) * SEQ + s0 + threadIdx.x];
    __syncthreads();
    for (int i = threadIdx.y; i < 32; i += 8)
        dst[((size_t)b * SEQ + s0 + i) * HIDDEN + c0 + threadIdx.x] = tile[threadIdx.x][i];
}

// ---------------------------------------------------------------------------
extern "C" void kernel_launch(void* const* d_in, const int* in_sizes, int n_in,
                              void* d_out, int out_size, void* d_ws, size_t ws_size,
                              hipStream_t stream) {
    const float* x  = (const float*)d_in[0];
    const float* Wq = (const float*)d_in[1];
    const float* bq = (const float*)d_in[2];
    const float* Wk = (const float*)d_in[3];
    const float* bk = (const float*)d_in[4];
    const float* Wv = (const float*)d_in[5];
    const float* bv = (const float*)d_in[6];
    const float* Wo = (const float*)d_in[7];
    const float* bo = (const float*)d_in[8];
    float* out = (float*)d_out;

    char* ws = (char*)d_ws;
    unsigned short* xb   = (unsigned short*)(ws);                             // 32 MB (reused as attn bf16)
    unsigned short* Wblk = (unsigned short*)(ws + (size_t)32 * 1024 * 1024);  // 8 MB (4 blocked mats)
    float* qbuf          = (float*)(ws + (size_t)40 * 1024 * 1024);           // 64 MB [B*C][S]
    float* kbuf          = (float*)(ws + (size_t)104 * 1024 * 1024);          // 64 MB
    float* vbuf          = (float*)(ws + (size_t)168 * 1024 * 1024);          // 64 MB
    unsigned short* outc = (unsigned short*)(ws + (size_t)232 * 1024 * 1024); // 32 MB [B*C][S]

    cvt_x_kernel<<<ROWS * HIDDEN / 256, 256, 0, stream>>>(x, xb);
    cvt_w_kernel<<<dim3(HIDDEN * HIDDEN / 256, 4), 256, 0, stream>>>(Wq, Wk, Wv, Wo, Wblk);

    // Q/K/V projections (blockIdx.z selects matrix), channel-major fp32 outputs
    gemm_wmma_kernel<0><<<dim3(HIDDEN / 128, ROWS / 128, 3), 256, 0, stream>>>(
        xb, Wblk, bq, bk, bv, qbuf, kbuf, vbuf);

    // Spectral mixing, one channel per workgroup, fully coalesced
    fftconv_kernel<<<BATCH * HIDDEN, 256, 0, stream>>>(qbuf, kbuf, vbuf, outc);

    // [B][C][S] -> [B][S][C] for the output projection
    transpose_kernel<<<dim3(SEQ / 32, HIDDEN / 32, BATCH), dim3(32, 8), 0, stream>>>(outc, xb);

    // Output projection -> fp32 d_out
    gemm_wmma_kernel<1><<<dim3(HIDDEN / 128, ROWS / 128, 1), 256, 0, stream>>>(
        xb, Wblk + (size_t)3 * HIDDEN * HIDDEN, bo, nullptr, nullptr, out, nullptr, nullptr);
}
